// PatchAdapterLayer_18442589569297
// MI455X (gfx1250) — matrix operand order repaired and checked by reading, not compiled, not run on hardware
//
#include <hip/hip_runtime.h>
#include <hip/hip_bf16.h>
#include <math.h>

#define DIMC   64
#define RANKC  32
#define NEXP   4
#define BATCH  4
#define HH     256
#define WWI    256
#define HWP    65536            // HH*WWI
#define NPIXI  262144           // BATCH*HWP

typedef __attribute__((ext_vector_type(16))) _Float16 v16h;
typedef __attribute__((ext_vector_type(8)))  float    v8f;
typedef __attribute__((ext_vector_type(2)))  _Float16 h2;
typedef __attribute__((ext_vector_type(4)))  unsigned v4u;
typedef __attribute__((ext_vector_type(4)))  int      v4i;
typedef __attribute__((address_space(1)))    v4i      as1_v4i;
typedef __attribute__((address_space(3)))    v4i      as3_v4i;

#if __has_builtin(__builtin_amdgcn_global_load_async_to_lds_b128) && \
    __has_builtin(__builtin_amdgcn_s_wait_asynccnt)
#define USE_ASYNC_LDS 1
#endif

__device__ __forceinline__ void async_ld128(const _Float16* g, _Float16* l) {
#ifdef USE_ASYNC_LDS
  as1_v4i* gp = (as1_v4i*)(size_t)g;              // 64-bit global address
  as3_v4i* lp = (as3_v4i*)(unsigned)(size_t)l;    // low 32 bits = LDS offset
  __builtin_amdgcn_global_load_async_to_lds_b128(gp, lp, 0, 0);
#else
  *(v4u*)l = *(const v4u*)g;
#endif
}

__device__ __forceinline__ void wait_async_lds() {
#ifdef USE_ASYNC_LDS
  __builtin_amdgcn_s_wait_asynccnt(0);
#endif
}

__device__ __forceinline__ v8f wmma16(v16h a, v16h b, v8f c) {
  return __builtin_amdgcn_wmma_f32_16x16x32_f16(
      false, a, false, b, (short)0, c, false, false);
}

// A fragment: 16 pixels x 32 channels from NHWC f16 (global or LDS).
// ISA 16-bit A layout: lanes0-15 hold K={0..7,16..23}, lanes16-31 K={8..15,24..31}.
__device__ __forceinline__ v16h load_afrag_h(const _Float16* src, int C,
                                             long pix0, int kbase, int lane) {
  const int m = lane & 15, hf = (lane >> 4) & 1;
  const _Float16* p = src + (pix0 + m) * (long)C + kbase;
  v16h a;
#pragma unroll
  for (int j = 0; j < 8; ++j) a[j] = p[hf * 8 + j];
#pragma unroll
  for (int j = 0; j < 8; ++j) a[8 + j] = p[16 + hf * 8 + j];
  return a;
}

__device__ __forceinline__ v16h load_afrag_f(const float* src, int C,
                                             long pix0, int kbase, int lane) {
  const int m = lane & 15, hf = (lane >> 4) & 1;
  const float* p = src + (pix0 + m) * (long)C + kbase;
  v16h a;
#pragma unroll
  for (int j = 0; j < 8; ++j) a[j] = (_Float16)p[hf * 8 + j];
#pragma unroll
  for (int j = 0; j < 8; ++j) a[8 + j] = (_Float16)p[16 + hf * 8 + j];
  return a;
}

// B fragment pre-packed in lane order: one 32B contiguous load per lane.
__device__ __forceinline__ v16h load_bfrag_p(const _Float16* wp, int frag, int lane) {
  return *(const v16h*)(wp + ((size_t)frag << 9) + ((size_t)lane << 4));
}

// Packed-weight fragment segment bases (fragment indices)
#define FR_P0   0     // e*4 + nt*2 + kt            (16)
#define FR_P1   16    // e*4 + nt*2 + kt            (16)
#define FR_Q    32    // e*2 + nt                   (8)
#define FR_KV   40    // e*4 + nt                   (16)
#define FR_FP   56    // e*2 + nt                   (8)
#define FR_P2   64    // e*4 + nt                   (16)
#define FR_OUT  80    // nt*2 + kt                  (8)
#define FR_TOTAL 88

// ---------------- W0: pack all weight tables into f16 WMMA-B fragment order -------
__global__ void k_wprep(const float* __restrict__ p0w, const float* __restrict__ p1w,
                        const float* __restrict__ qw, const float* __restrict__ kvw,
                        const float* __restrict__ fpw, const float* __restrict__ p2w,
                        const float* __restrict__ outw, _Float16* __restrict__ wpack) {
  int t = blockIdx.x * blockDim.x + threadIdx.x;       // FR_TOTAL*512
  int frag = t >> 9, r = t & 511;
  int lane = r >> 4, j = r & 15;
  int m = lane & 15, hf = (lane >> 4) & 1;
  const float* src; int inC, n0, kb;
  if (frag < 16)      { int f = frag;      src = p0w + (f >> 2) * 2048; inC = 64; n0 = ((f >> 1) & 1) * 16; kb = (f & 1) * 32; }
  else if (frag < 32) { int f = frag - 16; src = p1w + (f >> 2) * 2048; inC = 64; n0 = ((f >> 1) & 1) * 16; kb = (f & 1) * 32; }
  else if (frag < 40) { int f = frag - 32; src = qw  + (f >> 1) * 1024; inC = 32; n0 = (f & 1) * 16;        kb = 0; }
  else if (frag < 56) { int f = frag - 40; src = kvw + (f >> 2) * 2048; inC = 32; n0 = (f & 3) * 16;        kb = 0; }
  else if (frag < 64) { int f = frag - 56; src = fpw + (f >> 1) * 1024; inC = 32; n0 = (f & 1) * 16;        kb = 0; }
  else if (frag < 80) { int f = frag - 64; src = p2w + (f >> 2) * 2048; inC = 32; n0 = (f & 3) * 16;        kb = 0; }
  else                { int f = frag - 80; src = outw;                  inC = 64; n0 = (f >> 1) * 16;       kb = (f & 1) * 32; }
  wpack[t] = (_Float16)src[(size_t)(n0 + m) * inC + kb + hf * 8 + (j & 7) + (j >> 3) * 16];
}

// ---------------- K0: router softmax + top-1 gates ----------------
__global__ void k_router(const float* __restrict__ x, const float* __restrict__ rw,
                         const float* __restrict__ rb, float* __restrict__ gates) {
  int pix = blockIdx.x * blockDim.x + threadIdx.x;
  int b = pix >> 16, off = pix & (HWP - 1);
  const float* xp = x + (size_t)b * DIMC * HWP + off;
  float lg[NEXP];
#pragma unroll
  for (int e = 0; e < NEXP; ++e) lg[e] = rb[e];
  for (int c = 0; c < DIMC; ++c) {
    float xv = xp[(size_t)c * HWP];
#pragma unroll
    for (int e = 0; e < NEXP; ++e) lg[e] += xv * rw[e * DIMC + c];
  }
  float mx = lg[0]; int am = 0;
#pragma unroll
  for (int e = 1; e < NEXP; ++e) if (lg[e] > mx) { mx = lg[e]; am = e; }
  float den = 0.f;
#pragma unroll
  for (int e = 0; e < NEXP; ++e) den += expf(lg[e] - mx);
  float p = 1.f / den;
#pragma unroll
  for (int e = 0; e < NEXP; ++e)
    gates[(size_t)e * NPIXI + pix] = (e == am) ? p : 0.f;
}

// ---------------- K1: NCHW fp32 -> NHWC f16 pack; zero combined ----------------
__global__ void k_pack(const float* __restrict__ x, const float* __restrict__ s,
                       _Float16* __restrict__ xh, _Float16* __restrict__ sh,
                       float* __restrict__ combined) {
  long i = (long)blockIdx.x * blockDim.x + threadIdx.x;
  long b = i >> 22;
  long rem = i & 4194303;
  long c = rem >> 16, off = rem & (HWP - 1);
  long pix = (b << 16) + off;
  xh[pix * DIMC + c] = (_Float16)x[i];
  sh[pix * DIMC + c] = (_Float16)s[i];
  combined[i] = 0.f;
}

// ---------------- K2: h1 = mask*(p0·x), gsil = silu(mask*(p1·shared)) ----------------
__global__ void k_front(const _Float16* __restrict__ xh, const _Float16* __restrict__ sh,
                        const float* __restrict__ gates, const _Float16* __restrict__ wp,
                        _Float16* __restrict__ h1, _Float16* __restrict__ gsil, int e) {
  const int lane = threadIdx.x & 31, wave = threadIdx.x >> 5;
  const long pix0 = ((long)blockIdx.x * 8 + wave) * 16;
  const float* gate = gates + (size_t)e * NPIXI;
  v16h ax0 = load_afrag_h(xh, DIMC, pix0, 0, lane);
  v16h ax1 = load_afrag_h(xh, DIMC, pix0, 32, lane);
  v16h as0 = load_afrag_h(sh, DIMC, pix0, 0, lane);
  v16h as1 = load_afrag_h(sh, DIMC, pix0, 32, lane);
  const int m = lane & 15, hf = (lane >> 4) & 1;
#pragma unroll
  for (int nt = 0; nt < 2; ++nt) {
    v8f acc0 = {}; v8f acc1 = {};
    acc0 = wmma16(ax0, load_bfrag_p(wp, FR_P0 + e * 4 + nt * 2 + 0, lane), acc0);
    acc0 = wmma16(ax1, load_bfrag_p(wp, FR_P0 + e * 4 + nt * 2 + 1, lane), acc0);
    acc1 = wmma16(as0, load_bfrag_p(wp, FR_P1 + e * 4 + nt * 2 + 0, lane), acc1);
    acc1 = wmma16(as1, load_bfrag_p(wp, FR_P1 + e * 4 + nt * 2 + 1, lane), acc1);
#pragma unroll
    for (int r = 0; r < 8; ++r) {
      long pix = pix0 + r + hf * 8;
      float msk = gate[pix];
      float hv = acc0[r] * msk;
      float gv = acc1[r] * msk;
      gv = gv / (1.f + expf(-gv));
      h1[pix * RANKC + nt * 16 + m] = (_Float16)hv;
      gsil[pix * RANKC + nt * 16 + m] = (_Float16)gv;
    }
  }
}

// ---------------- K3: q0 = q_w·h1, kv0 = kv_w·h1 ----------------
__global__ void k_qkv(const _Float16* __restrict__ h1, const _Float16* __restrict__ wp,
                      _Float16* __restrict__ q0, _Float16* __restrict__ kv0, int e) {
  const int lane = threadIdx.x & 31, wave = threadIdx.x >> 5;
  const long pix0 = ((long)blockIdx.x * 8 + wave) * 16;
  const int m = lane & 15, hf = (lane >> 4) & 1;
  v16h a = load_afrag_h(h1, RANKC, pix0, 0, lane);
#pragma unroll
  for (int nt = 0; nt < 2; ++nt) {
    v8f acc = {};
    acc = wmma16(a, load_bfrag_p(wp, FR_Q + e * 2 + nt, lane), acc);
#pragma unroll
    for (int r = 0; r < 8; ++r)
      q0[(pix0 + r + hf * 8) * RANKC + nt * 16 + m] = (_Float16)acc[r];
  }
#pragma unroll
  for (int nt = 0; nt < 4; ++nt) {
    v8f acc = {};
    acc = wmma16(a, load_bfrag_p(wp, FR_KV + e * 4 + nt, lane), acc);
#pragma unroll
    for (int r = 0; r < 8; ++r)
      kv0[(pix0 + r + hf * 8) * DIMC + nt * 16 + m] = (_Float16)acc[r];
  }
}

// ---------------- K4a: LDS-tiled depthwise 3x3 (32ch), packed f16 pairs ----------------
__global__ void k_dw3(const _Float16* __restrict__ q0, const float* __restrict__ dww,
                      const float* __restrict__ dwb, _Float16* __restrict__ q, int e) {
  __shared__ _Float16 tile[18 * 18][RANKC];            // 20.7 KB
  const int tid = threadIdx.x;
  const int blk = blockIdx.x;                           // B*16*16
  const int b = blk >> 8, ty = (blk >> 4) & 15, tx = blk & 15;
  const int y0 = ty * 16, x0 = tx * 16;
  for (int t = tid; t < 324 * 2; t += 256) {
    int pp = t >> 1, ch = t & 1;
    int y = y0 - 1 + pp / 18, x = x0 - 1 + pp % 18;
    if (y >= 0 && y < HH && x >= 0 && x < WWI) {
      long gp = ((long)b << 16) + (y << 8) + x;
      async_ld128(q0 + gp * RANKC + ch * 8, &tile[pp][ch * 8]);
    } else {
      v4u z = {0u, 0u, 0u, 0u};
      *(v4u*)&tile[pp][ch * 8] = z;
    }
  }
  wait_async_lds();
  __syncthreads();

  const int cp = tid & 15, pg = tid >> 4;               // 16 ch-pairs, 16 pixel groups
  const int c0 = cp * 2;
  h2 wv[9];
#pragma unroll
  for (int k = 0; k < 9; ++k) {
    wv[k][0] = (_Float16)dww[((size_t)e * RANKC + c0) * 9 + k];
    wv[k][1] = (_Float16)dww[((size_t)e * RANKC + c0 + 1) * 9 + k];
  }
  float b0 = dwb[e * RANKC + c0], b1 = dwb[e * RANKC + c0 + 1];
#pragma unroll
  for (int it = 0; it < 16; ++it) {
    int pi = pg * 16 + it;
    int py = pi >> 4, px = pi & 15;
    h2 acc = {(_Float16)0.f, (_Float16)0.f};
#pragma unroll
    for (int dy = 0; dy < 3; ++dy)
#pragma unroll
      for (int dx = 0; dx < 3; ++dx)
        acc += wv[dy * 3 + dx] * (*(const h2*)&tile[(py + dy) * 18 + px + dx][c0]);
    long gp = ((long)b << 16) + ((y0 + py) << 8) + x0 + px;
    h2 o; o[0] = (_Float16)((float)acc[0] + b0); o[1] = (_Float16)((float)acc[1] + b1);
    *(h2*)&q[gp * RANKC + c0] = o;
  }
}

// ---------------- K4b: LDS-tiled depthwise 7x7 (64ch) -> k,v ----------------
__global__ void k_dw7(const _Float16* __restrict__ kv0, const float* __restrict__ dww,
                      const float* __restrict__ dwb, _Float16* __restrict__ kb,
                      _Float16* __restrict__ vb, int e) {
  __shared__ _Float16 tile[22 * 22][DIMC];              // 60.5 KB
  const int tid = threadIdx.x;
  const int blk = blockIdx.x;
  const int b = blk >> 8, ty = (blk >> 4) & 15, tx = blk & 15;
  const int y0 = ty * 16, x0 = tx * 16;
  for (int t = tid; t < 484 * 4; t += 256) {
    int pp = t >> 2, ch = t & 3;
    int y = y0 - 3 + pp / 22, x = x0 - 3 + pp % 22;
    if (y >= 0 && y < HH && x >= 0 && x < WWI) {
      long gp = ((long)b << 16) + (y << 8) + x;
      async_ld128(kv0 + gp * DIMC + ch * 8, &tile[pp][ch * 8]);
    } else {
      v4u z = {0u, 0u, 0u, 0u};
      *(v4u*)&tile[pp][ch * 8] = z;
    }
  }
  wait_async_lds();
  __syncthreads();

  const int cp = tid & 31, pg = tid >> 5;               // 32 ch-pairs, 8 pixel groups
  const int c0 = cp * 2;
  h2 wv[49];
#pragma unroll
  for (int k = 0; k < 49; ++k) {
    wv[k][0] = (_Float16)dww[((size_t)e * DIMC + c0) * 49 + k];
    wv[k][1] = (_Float16)dww[((size_t)e * DIMC + c0 + 1) * 49 + k];
  }
  float b0 = dwb[e * DIMC + c0], b1 = dwb[e * DIMC + c0 + 1];
  for (int it = 0; it < 32; ++it) {
    int pi = pg * 32 + it;
    int py = pi >> 4, px = pi & 15;
    h2 acc = {(_Float16)0.f, (_Float16)0.f};
#pragma unroll
    for (int dy = 0; dy < 7; ++dy)
#pragma unroll
      for (int dx = 0; dx < 7; ++dx)
        acc += wv[dy * 7 + dx] * (*(const h2*)&tile[(py + dy) * 22 + px + dx][c0]);
    long gp = ((long)b << 16) + ((y0 + py) << 8) + x0 + px;
    h2 o; o[0] = (_Float16)((float)acc[0] + b0); o[1] = (_Float16)((float)acc[1] + b1);
    if (cp < 16) *(h2*)&kb[gp * RANKC + c0] = o;
    else         *(h2*)&vb[gp * RANKC + (c0 - RANKC)] = o;
  }
}

// ---------------- K5: patch circconv + LN + fp GEMM + gate + p2 GEMM + combine ----------
__global__ void k_tail(const _Float16* __restrict__ qb, const _Float16* __restrict__ kbuf,
                       const _Float16* __restrict__ vb, const _Float16* __restrict__ gsil,
                       const float* __restrict__ gates, const _Float16* __restrict__ xh,
                       const float* __restrict__ lnw, const float* __restrict__ lnb,
                       const float* __restrict__ fpb, const _Float16* __restrict__ wp,
                       float* __restrict__ combined, int e) {
  __shared__ _Float16 sq[64][RANKC];
  __shared__ _Float16 sk[64][RANKC];
  __shared__ float    so[64][RANKC];
  __shared__ _Float16 sbody[64][RANKC];
  __shared__ _Float16 sh2[64][RANKC];

  const int tid = threadIdx.x;
  const int patch = blockIdx.x;
  const int b = patch >> 10;
  const int pr = (patch >> 5) & 31, pc = patch & 31;
  const int y0 = pr * 8, x0 = pc * 8;
  const float* gate = gates + (size_t)e * NPIXI;

  // phase 1: stage q,k patch into LDS via async loads: 2 arrays * 64 pix * 4 chunks
#pragma unroll
  for (int rep = 0; rep < 2; ++rep) {
    int t = tid + rep * 256;
    int arr = t >> 8, pp = (t >> 2) & 63, ch = t & 3;
    long gp = ((long)b << 16) + ((y0 + (pp >> 3)) << 8) + x0 + (pp & 7);
    const _Float16* gsrc = (arr ? kbuf : qb) + gp * RANKC + ch * 8;
    _Float16* ldst = arr ? &sk[pp][ch * 8] : &sq[pp][ch * 8];
    async_ld128(gsrc, ldst);
  }
  wait_async_lds();
  __syncthreads();

  // phase 2: 8x8 circular convolution, packed channel pairs
#pragma unroll
  for (int rep = 0; rep < 4; ++rep) {
    int t = tid + rep * 256;                            // 64 pp * 16 cpairs
    int pp = t >> 4, cp = t & 15;
    int u = pp >> 3, v = pp & 7;
    h2 acc = {(_Float16)0.f, (_Float16)0.f};
#pragma unroll
    for (int s = 0; s < 8; ++s) {
      int ku = (u - s) & 7;
#pragma unroll
      for (int t2 = 0; t2 < 8; ++t2) {
        int kv = (v - t2) & 7;
        acc += (*(const h2*)&sq[s * 8 + t2][cp * 2]) *
               (*(const h2*)&sk[ku * 8 + kv][cp * 2]);
      }
    }
    so[pp][cp * 2]     = (float)acc[0];
    so[pp][cp * 2 + 1] = (float)acc[1];
  }
  __syncthreads();

  // phase 3: per-pixel LayerNorm over 32 channels, * ln, * v
  if (tid < 64) {
    int pp = tid;
    long gp = ((long)b << 16) + ((y0 + (pp >> 3)) << 8) + x0 + (pp & 7);
    float mu = 0.f, m2 = 0.f;
#pragma unroll
    for (int c = 0; c < RANKC; ++c) { float t = so[pp][c]; mu += t; m2 += t * t; }
    mu *= (1.f / RANKC); m2 *= (1.f / RANKC);
    float rstd = rsqrtf(m2 - mu * mu + 1e-5f);
#pragma unroll
    for (int c = 0; c < RANKC; ++c) {
      float t = (so[pp][c] - mu) * rstd * lnw[e * RANKC + c] + lnb[e * RANKC + c];
      t *= (float)vb[gp * RANKC + c];
      sbody[pp][c] = (_Float16)t;
    }
  }
  __syncthreads();

  // phase 4: fp GEMM (32->32) + bias, * gsil; 8 wave-tiles
  const int lane = tid & 31, wave = tid >> 5;
  const int m = lane & 15, hf = (lane >> 4) & 1;
  {
    int mt = wave >> 1, nt = wave & 1;
    v16h a = load_afrag_h(&sbody[0][0], RANKC, mt * 16, 0, lane);
    v8f acc = {};
    acc = wmma16(a, load_bfrag_p(wp, FR_FP + e * 2 + nt, lane), acc);
    float bias = fpb[e * RANKC + nt * 16 + m];
#pragma unroll
    for (int r = 0; r < 8; ++r) {
      int pp = mt * 16 + r + hf * 8;
      long gp = ((long)b << 16) + ((y0 + (pp >> 3)) << 8) + x0 + (pp & 7);
      float g = (float)gsil[gp * RANKC + nt * 16 + m];
      sh2[pp][nt * 16 + m] = (_Float16)((acc[r] + bias) * g);
    }
  }
  __syncthreads();

  // phase 5: p2 GEMM (32->64); combined += mask*p2out + mask^2 * x
#pragma unroll
  for (int rep = 0; rep < 2; ++rep) {
    int t = wave * 2 + rep;
    int mt = t >> 2, nt = t & 3;
    v16h a = load_afrag_h(&sh2[0][0], RANKC, mt * 16, 0, lane);
    v8f acc = {};
    acc = wmma16(a, load_bfrag_p(wp, FR_P2 + e * 4 + nt, lane), acc);
#pragma unroll
    for (int r = 0; r < 8; ++r) {
      int pp = mt * 16 + r + hf * 8;
      long gp = ((long)b << 16) + ((y0 + (pp >> 3)) << 8) + x0 + (pp & 7);
      int n = nt * 16 + m;
      float msk = gate[gp];
      float xv = (float)xh[gp * DIMC + n];
      combined[gp * DIMC + n] += msk * acc[r] + msk * msk * xv;
    }
  }
}

// ---------------- K6: out = out_w · combined + out_b, NCHW fp32 ----------------
__global__ void k_out(const float* __restrict__ combined, const _Float16* __restrict__ wp,
                      const float* __restrict__ outb, float* __restrict__ out) {
  const int lane = threadIdx.x & 31, wave = threadIdx.x >> 5;
  const long pix0 = ((long)blockIdx.x * 8 + wave) * 16;
  const int m = lane & 15, hf = (lane >> 4) & 1;
  v16h a0 = load_afrag_f(combined, DIMC, pix0, 0, lane);
  v16h a1 = load_afrag_f(combined, DIMC, pix0, 32, lane);
#pragma unroll
  for (int nt = 0; nt < 4; ++nt) {
    v8f acc = {};
    acc = wmma16(a0, load_bfrag_p(wp, FR_OUT + nt * 2 + 0, lane), acc);
    acc = wmma16(a1, load_bfrag_p(wp, FR_OUT + nt * 2 + 1, lane), acc);
    int n = nt * 16 + m;
    float bias = outb[n];
#pragma unroll
    for (int r = 0; r < 8; ++r) {
      long pix = pix0 + r + hf * 8;
      long bb = pix >> 16, off = pix & (HWP - 1);
      out[(bb * DIMC + n) * (long)HWP + off] = acc[r] + bias;
    }
  }
}

extern "C" void kernel_launch(void* const* d_in, const int* in_sizes, int n_in,
                              void* d_out, int out_size, void* d_ws, size_t ws_size,
                              hipStream_t stream) {
  (void)in_sizes; (void)n_in; (void)out_size; (void)ws_size;
  const float* x    = (const float*)d_in[0];
  const float* shr  = (const float*)d_in[1];
  const float* rw   = (const float*)d_in[2];
  const float* rb   = (const float*)d_in[3];
  const float* p0w  = (const float*)d_in[4];
  const float* p1w  = (const float*)d_in[5];
  const float* p2w  = (const float*)d_in[6];
  const float* qw   = (const float*)d_in[7];
  const float* qdww = (const float*)d_in[8];
  const float* qdwb = (const float*)d_in[9];
  const float* kvw  = (const float*)d_in[10];
  const float* kvdww= (const float*)d_in[11];
  const float* kvdwb= (const float*)d_in[12];
  const float* lnw  = (const float*)d_in[13];
  const float* lnb  = (const float*)d_in[14];
  const float* fpw  = (const float*)d_in[15];
  const float* fpb  = (const float*)d_in[16];
  const float* outw = (const float*)d_in[17];
  const float* outb = (const float*)d_in[18];
  float* out = (float*)d_out;

  char* ws = (char*)d_ws;
  size_t off = 0;
  auto wsAlloc = [&](size_t bytes) -> char* {
    char* p = ws + off; off += (bytes + 255) & ~(size_t)255; return p;
  };
  const size_t NP = (size_t)NPIXI;
  float*    gates    = (float*)   wsAlloc(NEXP * NP * 4);
  _Float16* xh       = (_Float16*)wsAlloc(NP * DIMC * 2);
  _Float16* sh       = (_Float16*)wsAlloc(NP * DIMC * 2);
  float*    combined = (float*)   wsAlloc(NP * DIMC * 4);
  _Float16* h1       = (_Float16*)wsAlloc(NP * RANKC * 2);
  _Float16* gsil     = (_Float16*)wsAlloc(NP * RANKC * 2);
  _Float16* q0       = (_Float16*)wsAlloc(NP * RANKC * 2);
  _Float16* kv0      = (_Float16*)wsAlloc(NP * DIMC * 2);
  _Float16* qb       = (_Float16*)wsAlloc(NP * RANKC * 2);
  _Float16* kb       = (_Float16*)wsAlloc(NP * RANKC * 2);
  _Float16* vb       = (_Float16*)wsAlloc(NP * RANKC * 2);
  _Float16* wpack    = (_Float16*)wsAlloc(FR_TOTAL * 512 * 2);

  k_wprep<<<FR_TOTAL * 2, 256, 0, stream>>>(p0w, p1w, qw, kvw, fpw, p2w, outw, wpack);
  k_router<<<NPIXI / 256, 256, 0, stream>>>(x, rw, rb, gates);
  k_pack<<<NPIXI * (DIMC / 256), 256, 0, stream>>>(x, shr, xh, sh, combined);

  for (int e = 0; e < NEXP; ++e) {
    k_front<<<NPIXI / 128, 256, 0, stream>>>(xh, sh, gates, wpack, h1, gsil, e);
    k_qkv<<<NPIXI / 128, 256, 0, stream>>>(h1, wpack, q0, kv0, e);
    k_dw3<<<BATCH * 16 * 16, 256, 0, stream>>>(q0, qdww, qdwb, qb, e);
    k_dw7<<<BATCH * 16 * 16, 256, 0, stream>>>(kv0, kvdww, kvdwb, kb, vb, e);
    k_tail<<<BATCH * 32 * 32, 256, 0, stream>>>(qb, kb, vb, gsil, gates, xh,
                                                lnw, lnb, fpb, wpack, combined, e);
  }
  k_out<<<NPIXI / 128, 256, 0, stream>>>(combined, wpack, outb, out);
}
